// GAT_1211180778444
// MI455X (gfx1250) — matrix operand order repaired and checked by reading, not compiled, not run on hardware
//
#include <hip/hip_runtime.h>
#include <hip/hip_bf16.h>

typedef __attribute__((ext_vector_type(16))) _Float16 v16h;
typedef __attribute__((ext_vector_type(8)))  float    v8f;

#define NN       100000
#define IN_DIM   128
#define HID      64
#define NEG_SLOPE 0.2f

// ---------- helpers ----------
__device__ __forceinline__ float lrelu(float x) { return x > 0.0f ? x : NEG_SLOPE * x; }

// monotonic unsigned encoding of float (identity element for max == 0u)
__device__ __forceinline__ unsigned encf(float x) {
  unsigned u = __float_as_uint(x);
  return (u & 0x80000000u) ? ~u : (u | 0x80000000u);
}
__device__ __forceinline__ float decf(unsigned e) {
  return (e & 0x80000000u) ? __uint_as_float(e & 0x7FFFFFFFu) : __uint_as_float(~e);
}

__device__ __forceinline__ void edge_sd(const long long* ei, long long Eraw,
                                        long long e, long long& s, long long& d) {
  if (e < Eraw) { s = ei[e]; d = ei[Eraw + e]; }
  else          { s = e - Eraw; d = e - Eraw; }   // self-loop
}

// ---------- zero-fill scratch ----------
__global__ void k_zero(float* __restrict__ p, long long cnt) {
  long long i = blockIdx.x * (long long)blockDim.x + threadIdx.x;
  long long stride = (long long)gridDim.x * blockDim.x;
  for (; i < cnt; i += stride) p[i] = 0.0f;
}

// ---------- layer 1 GEMM: H1 = x @ W1  (WMMA f16, f32 accum) ----------
__global__ __launch_bounds__(256) void k_gemm1(const float* __restrict__ x,
                                               const float* __restrict__ W1,
                                               float* __restrict__ H1) {
  const int ntiles = NN / 16;                 // 6250, exact
  int lane = threadIdx.x & 31;
  int wave = threadIdx.x >> 5;
  int tile = blockIdx.x * 8 + wave;
  if (tile >= ntiles) return;

  int l15 = lane & 15;
  int kg  = lane >> 4;                        // lane group 0/1
  int row = tile * 16 + l15;

  v8f z = {0.f,0.f,0.f,0.f,0.f,0.f,0.f,0.f};
  v8f acc[4] = {z, z, z, z};

  for (int kk = 0; kk < 4; ++kk) {            // K = 128 in 4 steps of 32
    // A fragment: lane holds row M=l15; halves 0-7 -> K=kk*32+kg*8+j,
    //             halves 8-15 -> K=kk*32+16+kg*8+j
    v16h a;
    {
      const float* ap = x + (size_t)row * IN_DIM + kk * 32 + kg * 8;
      float4 f0 = *(const float4*)(ap);
      float4 f1 = *(const float4*)(ap + 4);
      float4 f2 = *(const float4*)(ap + 16);
      float4 f3 = *(const float4*)(ap + 20);
      a[0]=(_Float16)f0.x; a[1]=(_Float16)f0.y; a[2]=(_Float16)f0.z; a[3]=(_Float16)f0.w;
      a[4]=(_Float16)f1.x; a[5]=(_Float16)f1.y; a[6]=(_Float16)f1.z; a[7]=(_Float16)f1.w;
      a[8]=(_Float16)f2.x; a[9]=(_Float16)f2.y; a[10]=(_Float16)f2.z; a[11]=(_Float16)f2.w;
      a[12]=(_Float16)f3.x; a[13]=(_Float16)f3.y; a[14]=(_Float16)f3.z; a[15]=(_Float16)f3.w;
    }
    #pragma unroll
    for (int nt = 0; nt < 4; ++nt) {
      // B fragment: lane l15 = column N; halves 0-7 -> K = kk*32 + kg*8 + j,
      // halves 8-15 -> K = kk*32 + 16 + kg*8 + (j-8). W1 is row-major [K=128, N=64].
      v16h b;
      const float* bp = W1 + (size_t)(kk * 32 + kg * 8) * HID + nt * 16 + l15;
      #pragma unroll
      for (int j = 0; j < 16; ++j) b[j] = (_Float16)bp[(size_t)(j < 8 ? j : j + 8) * HID];
      acc[nt] = __builtin_amdgcn_wmma_f32_16x16x32_f16(
          false, a, false, b, (short)0, acc[nt], false, false);
    }
  }
  // C/D layout: lane<16 -> (M=r, N=l15); lane>=16 -> (M=r+8, N=l15)
  int mofs = kg * 8;
  #pragma unroll
  for (int nt = 0; nt < 4; ++nt) {
    #pragma unroll
    for (int r = 0; r < 8; ++r) {
      H1[(size_t)(tile * 16 + mofs + r) * HID + nt * 16 + l15] = acc[nt][r];
    }
  }
}

// ---------- layer 1 per-node attention logits ----------
__global__ void k_alpha1(const float* __restrict__ H1,
                         const float* __restrict__ a_src,
                         const float* __restrict__ a_dst,
                         float* __restrict__ als, float* __restrict__ ald) {
  int n = blockIdx.x * blockDim.x + threadIdx.x;
  if (n >= NN) return;
  const float* hp = H1 + (size_t)n * HID;
  #pragma unroll
  for (int h = 0; h < 8; ++h) {
    float ss = 0.f, sd = 0.f;
    #pragma unroll
    for (int o = 0; o < 8; ++o) {
      float v = hp[h * 8 + o];
      ss += v * a_src[h * 8 + o];
      sd += v * a_dst[h * 8 + o];
    }
    als[(size_t)n * 8 + h] = ss;
    ald[(size_t)n * 8 + h] = sd;
  }
}

// ---------- layer 1 segment max ----------
__global__ void k_l1_max(const long long* __restrict__ ei, long long Eraw, long long Etot,
                         const float* __restrict__ als, const float* __restrict__ ald,
                         unsigned* __restrict__ M1) {
  long long e = blockIdx.x * (long long)blockDim.x + threadIdx.x;
  if (e >= Etot) return;
  long long s, d; edge_sd(ei, Eraw, e, s, d);
  const float* ps = als + s * 8;
  const float* pd = ald + d * 8;
  unsigned* pm = M1 + d * 8;
  #pragma unroll
  for (int h = 0; h < 8; ++h) atomicMax(pm + h, encf(lrelu(ps[h] + pd[h])));
}

// ---------- layer 1 segment exp-sum ----------
__global__ void k_l1_sum(const long long* __restrict__ ei, long long Eraw, long long Etot,
                         const float* __restrict__ als, const float* __restrict__ ald,
                         const unsigned* __restrict__ M1, float* __restrict__ D1) {
  long long e = blockIdx.x * (long long)blockDim.x + threadIdx.x;
  if (e >= Etot) return;
  long long s, d; edge_sd(ei, Eraw, e, s, d);
  const float* ps = als + s * 8;
  const float* pd = ald + d * 8;
  #pragma unroll
  for (int h = 0; h < 8; ++h) {
    float v = lrelu(ps[h] + pd[h]);
    atomicAdd(D1 + d * 8 + h, __expf(v - decf(M1[d * 8 + h])));
  }
}

// ---------- layer 1 normalize + aggregate (thread per edge x head) ----------
__global__ void k_l1_agg(const long long* __restrict__ ei, long long Eraw, long long Etot,
                         const float* __restrict__ als, const float* __restrict__ ald,
                         const unsigned* __restrict__ M1, const float* __restrict__ D1,
                         const float* __restrict__ H1, float* __restrict__ AGG) {
  long long t = blockIdx.x * (long long)blockDim.x + threadIdx.x;
  if (t >= Etot * 8) return;
  long long e = t >> 3;
  int h = (int)(t & 7);
  long long s, d; edge_sd(ei, Eraw, e, s, d);
  float v = lrelu(als[s * 8 + h] + ald[d * 8 + h]);
  float alpha = __expf(v - decf(M1[d * 8 + h])) / D1[d * 8 + h];
  const float4* hp = (const float4*)(H1 + (size_t)s * HID + h * 8);
  float4 h0 = hp[0], h1 = hp[1];
  float* op = AGG + (size_t)d * HID + h * 8;
  atomicAdd(op + 0, h0.x * alpha); atomicAdd(op + 1, h0.y * alpha);
  atomicAdd(op + 2, h0.z * alpha); atomicAdd(op + 3, h0.w * alpha);
  atomicAdd(op + 4, h1.x * alpha); atomicAdd(op + 5, h1.y * alpha);
  atomicAdd(op + 6, h1.z * alpha); atomicAdd(op + 7, h1.w * alpha);
}

// ---------- bias + relu + layer-2 GEMV + layer-2 logits ----------
__global__ void k_h2(const float* __restrict__ AGG, const float* __restrict__ b1,
                     const float* __restrict__ W2,
                     const float* __restrict__ a_src2, const float* __restrict__ a_dst2,
                     float* __restrict__ h2, float* __restrict__ al2s,
                     float* __restrict__ al2d) {
  int n = blockIdx.x * blockDim.x + threadIdx.x;
  if (n >= NN) return;
  const float* ap = AGG + (size_t)n * HID;
  float acc = 0.f;
  #pragma unroll
  for (int c = 0; c < HID; ++c) {
    float v = ap[c] + b1[c];
    v = v > 0.f ? v : 0.f;
    acc += v * W2[c];
  }
  h2[n]   = acc;
  al2s[n] = acc * a_src2[0];
  al2d[n] = acc * a_dst2[0];
}

// ---------- output init with bias b2 ----------
__global__ void k_out_init(float* __restrict__ out, const float* __restrict__ b2) {
  int n = blockIdx.x * blockDim.x + threadIdx.x;
  if (n < NN) out[n] = b2[0];
}

// ---------- layer 2 edge passes (heads = 1, out = 1) ----------
__global__ void k_l2_max(const long long* __restrict__ ei, long long Eraw, long long Etot,
                         const float* __restrict__ als, const float* __restrict__ ald,
                         unsigned* __restrict__ M2) {
  long long e = blockIdx.x * (long long)blockDim.x + threadIdx.x;
  if (e >= Etot) return;
  long long s, d; edge_sd(ei, Eraw, e, s, d);
  atomicMax(M2 + d, encf(lrelu(als[s] + ald[d])));
}

__global__ void k_l2_sum(const long long* __restrict__ ei, long long Eraw, long long Etot,
                         const float* __restrict__ als, const float* __restrict__ ald,
                         const unsigned* __restrict__ M2, float* __restrict__ D2) {
  long long e = blockIdx.x * (long long)blockDim.x + threadIdx.x;
  if (e >= Etot) return;
  long long s, d; edge_sd(ei, Eraw, e, s, d);
  float v = lrelu(als[s] + ald[d]);
  atomicAdd(D2 + d, __expf(v - decf(M2[d])));
}

__global__ void k_l2_agg(const long long* __restrict__ ei, long long Eraw, long long Etot,
                         const float* __restrict__ als, const float* __restrict__ ald,
                         const unsigned* __restrict__ M2, const float* __restrict__ D2,
                         const float* __restrict__ h2, float* __restrict__ out) {
  long long e = blockIdx.x * (long long)blockDim.x + threadIdx.x;
  if (e >= Etot) return;
  long long s, d; edge_sd(ei, Eraw, e, s, d);
  float v = lrelu(als[s] + ald[d]);
  float alpha = __expf(v - decf(M2[d])) / D2[d];
  atomicAdd(out + d, h2[s] * alpha);
}

// ---------- host launch ----------
extern "C" void kernel_launch(void* const* d_in, const int* in_sizes, int n_in,
                              void* d_out, int out_size, void* d_ws, size_t ws_size,
                              hipStream_t stream) {
  const float*     x      = (const float*)d_in[0];
  const float*     W1     = (const float*)d_in[1];
  const float*     a_src1 = (const float*)d_in[2];
  const float*     a_dst1 = (const float*)d_in[3];
  const float*     b1     = (const float*)d_in[4];
  const float*     W2     = (const float*)d_in[5];
  const float*     a_src2 = (const float*)d_in[6];
  const float*     a_dst2 = (const float*)d_in[7];
  const float*     b2     = (const float*)d_in[8];
  const long long* ei     = (const long long*)d_in[9];

  long long Eraw = (long long)in_sizes[9] / 2;   // 3.2M
  long long Etot = Eraw + NN;                    // + self-loops

  // workspace layout (floats)
  float* ws   = (float*)d_ws;
  float* H1   = ws;                                  // N*64
  float* ALS1 = H1   + (size_t)NN * HID;             // N*8
  float* ALD1 = ALS1 + (size_t)NN * 8;               // N*8
  float* H2v  = ALD1 + (size_t)NN * 8;               // N
  float* AL2S = H2v  + NN;                           // N
  float* AL2D = AL2S + NN;                           // N
  // contiguous zero-init region:
  unsigned* M1 = (unsigned*)(AL2D + NN);             // N*8
  float*    D1 = (float*)(M1 + (size_t)NN * 8);      // N*8
  float*    AGG = D1 + (size_t)NN * 8;               // N*64
  unsigned* M2 = (unsigned*)(AGG + (size_t)NN * HID);// N
  float*    D2 = (float*)(M2 + NN);                  // N
  long long zero_cnt = (long long)NN * (8 + 8 + 64 + 1 + 1); // 82*N floats

  k_zero<<<2048, 256, 0, stream>>>((float*)M1, zero_cnt);

  // layer-1 GEMM: 6250 tiles, 8 tiles (waves) per 256-thread block
  k_gemm1<<<(6250 + 7) / 8, 256, 0, stream>>>(x, W1, H1);

  int nb_node = (NN + 255) / 256;
  k_alpha1<<<nb_node, 256, 0, stream>>>(H1, a_src1, a_dst1, ALS1, ALD1);

  int nb_edge = (int)((Etot + 255) / 256);
  k_l1_max<<<nb_edge, 256, 0, stream>>>(ei, Eraw, Etot, ALS1, ALD1, M1);
  k_l1_sum<<<nb_edge, 256, 0, stream>>>(ei, Eraw, Etot, ALS1, ALD1, M1, D1);

  int nb_eh = (int)((Etot * 8 + 255) / 256);
  k_l1_agg<<<nb_eh, 256, 0, stream>>>(ei, Eraw, Etot, ALS1, ALD1, M1, D1, H1, AGG);

  k_h2<<<nb_node, 256, 0, stream>>>(AGG, b1, W2, a_src2, a_dst2, H2v, AL2S, AL2D);

  float* out = (float*)d_out;
  k_out_init<<<nb_node, 256, 0, stream>>>(out, b2);
  k_l2_max<<<nb_edge, 256, 0, stream>>>(ei, Eraw, Etot, AL2S, AL2D, M2);
  k_l2_sum<<<nb_edge, 256, 0, stream>>>(ei, Eraw, Etot, AL2S, AL2D, M2, D2);
  k_l2_agg<<<nb_edge, 256, 0, stream>>>(ei, Eraw, Etot, AL2S, AL2D, M2, D2, H2v, out);
}
